// SudokuNetwork_24017457119837
// MI455X (gfx1250) — compile-verified, matrix operand order checked by reading, not run on hardware
//
#include <hip/hip_runtime.h>

typedef __attribute__((ext_vector_type(16))) _Float16 v16h;
typedef __attribute__((ext_vector_type(8)))  _Float16 v8h;
typedef __attribute__((ext_vector_type(8)))  float    v8f;

#define N9     9
#define CELLS  81
#define HID    100
#define BATCH  2048
#define BOARD  (CELLS * N9)   // 729

#define MP    96     // cells padded -> 6 tiles of 16
#define HP    112    // hidden padded -> 7 tiles of 16
#define KP2   128    // hidden padded -> 4 K=32 steps (GEMM2)
#define HSTR  136    // H2 row stride in halves (272B: 16B aligned, 68 DW -> spread banks)
#define LSTR  96     // logitsT row stride in floats

#define GZERO 248    // gsumP slot holding 0.0f
#define GONE  249    // gsumP slot holding 1.0f (bias feature)

// K index inside a 16x16x32 f16 A/B fragment for element e (0..15) of a lane
__device__ __forceinline__ int wmma_k_of(int e, bool hi) {
    int v = e >> 1, pos = e & 1;
    int k = (v < 4 ? v * 2 : 16 + (v - 4) * 2) + pos;
    return k + (hi ? 8 : 0);
}

__global__ __launch_bounds__(256)
void sudoku_solver_kernel(const float* __restrict__ x_in,
                          const float* __restrict__ W1,
                          const float* __restrict__ b1,
                          const float* __restrict__ W2,
                          const float* __restrict__ b2,
                          const int*   __restrict__ n_iters_p,
                          float* __restrict__ out_xpred,
                          float* __restrict__ out_x)
{
    __shared__ float    xL[BOARD];
    __shared__ float    xpredL[BOARD];
    __shared__ __attribute__((aligned(16))) float gsumP[256];            // 243 counts + const slots
    __shared__ __attribute__((aligned(32))) _Float16 W1TB[7 * 32 * 16];  // A frags: W1^T (+b1 at k=27)
    __shared__ __attribute__((aligned(32))) _Float16 W2B [4 * 32 * 16];  // B frags: W2 (+b2 at k=112)
    __shared__ __attribute__((aligned(32))) _Float16 H2[MP * HSTR];      // H2[cell][hidden], col112=1
    __shared__ __attribute__((aligned(16))) float    logitsT[16 * LSTR]; // logitsT[digit][cell]
    __shared__ unsigned long long bestKey;

    const int  tid  = threadIdx.x;
    const int  b    = blockIdx.x;
    const int  lane = tid & 31;
    const int  wave = tid >> 5;
    const int  n16  = lane & 15;
    const bool hi   = lane >= 16;

    // ---- one-time staging ----
    for (int i = tid; i < BOARD; i += 256) {
        float v = x_in[b * BOARD + i];
        xL[i] = v;
        xpredL[i] = v;
    }
    // A fragments of GEMM1: A[m=hidden][k] = W1[k][m] for k<27, b1[m] at k==27
    for (int i = tid; i < 7 * 32 * 16; i += 256) {
        int ht = i >> 9, ln = (i >> 4) & 31, e = i & 15;
        int k  = wmma_k_of(e, ln >= 16);
        int m  = ht * 16 + (ln & 15);
        float v = 0.f;
        if (m < HID) {
            if (k < 27)       v = W1[k * HID + m];
            else if (k == 27) v = b1[m];
        }
        W1TB[i] = (_Float16)v;
    }
    // B fragments of GEMM2: B[k=hidden][n] = W2[k][n] for k<100, b2[n] at k==112
    for (int i = tid; i < 4 * 32 * 16; i += 256) {
        int kc = i >> 9, ln = (i >> 4) & 31, e = i & 15;
        int k  = kc * 32 + wmma_k_of(e, ln >= 16);
        int n  = ln & 15;
        float v = 0.f;
        if (n < N9) {
            if (k < HID)       v = W2[k * N9 + n];
            else if (k == 112) v = b2[n];
        }
        W2B[i] = (_Float16)v;
    }
    // constant slots of gsumP (243..255): zeros, 1.0 at GONE
    if (tid < 16) gsumP[240 + tid] = (240 + tid == GONE) ? 1.f : 0.f;
    // H2 pad columns [112,128): bias-feature 1.0 at col 112, zero elsewhere
    for (int i = tid; i < MP * 16; i += 256) {
        int c = i >> 4, kk = 112 + (i & 15);
        H2[c * HSTR + kk] = (kk == 112) ? (_Float16)1.f : (_Float16)0.f;
    }
    if (tid == 0) bestKey = 0ull;
    const int iters = n_iters_p[0];

    // ---- precompute iteration-invariant gsum indices for the cons^T B fragment ----
    int gidx[16];
    if (wave < 6) {
        const int cell = wave * 16 + n16;
        #pragma unroll
        for (int e = 0; e < 16; ++e) {
            int k  = wmma_k_of(e, hi);
            int t2 = k / 9, d = k - t2 * 9;
            int g  = (t2 == 0) ? cell / 9
                   : (t2 == 1) ? cell % 9
                   :             (cell / 27) * 3 + (cell % 9) / 3;
            int idx = t2 * 81 + g * 9 + d;              // valid only when k<27
            idx = (k < 27) ? idx : ((k == 27) ? GONE : GZERO);
            gidx[e] = (cell < CELLS) ? idx : GZERO;
        }
    }
    __syncthreads();

    for (int it = 0; it < iters; ++it) {
        // ---- 1) group digit counts: gsumP[type*81 + group*9 + digit] ----
        if (tid < 243) {
            int tt = tid / 81, r = tid % 81, g = r / 9, d = r % 9;
            float s = 0.f;
            #pragma unroll
            for (int i = 0; i < 9; ++i) {
                int c;
                if (tt == 0)      c = g * 9 + i;
                else if (tt == 1) c = i * 9 + g;
                else              c = (g / 3) * 27 + (g % 3) * 3 + (i / 3) * 9 + (i % 3);
                s += xL[c * 9 + d];
            }
            gsumP[tid] = s;
        }
        __syncthreads();

        // ---- 2) GEMM1: H^T[112+1,96] = relu(W1aug^T[112,32] @ consaug^T[32,96]) ----
        if (wave < 6) {
            const int cell = wave * 16 + n16;
            // branchless cons^T B fragment (16 unconditional ds_load_b32)
            v16h bfrag;
            #pragma unroll
            for (int e = 0; e < 16; ++e)
                bfrag[e] = (_Float16)gsumP[gidx[e]];
            const int hoff = hi ? 8 : 0;
            for (int ht = 0; ht < 7; ++ht) {
                v16h afrag = *(const v16h*)&W1TB[(ht * 32 + lane) * 16];
                v8f acc = {};   // bias folded into K=27 -> inline zero C
                acc = __builtin_amdgcn_wmma_f32_16x16x32_f16(
                        false, afrag, false, bfrag, (short)0, acc, false, false);
                // D row = hidden, col = cell -> 8 contiguous halves of H2[cell]
                v8h hv;
                #pragma unroll
                for (int j = 0; j < 8; ++j)
                    hv[j] = (_Float16)fmaxf(acc[j], 0.f);
                *(v8h*)&H2[cell * HSTR + ht * 16 + hoff] = hv;
            }
        }
        __syncthreads();

        // ---- 3) GEMM2: logits[96,16] = H2aug[96,128] @ W2aug[128,16] ----
        if (wave < 6) {
            const int mt   = wave;
            const int cell = mt * 16 + n16;
            const int hoff = hi ? 8 : 0;
            v8f acc = {};   // bias folded into K=112 -> inline zero C
            #pragma unroll
            for (int kc = 0; kc < 4; ++kc) {
                const int base = cell * HSTR + kc * 32 + hoff;
                v8h lo  = *(const v8h*)&H2[base];
                v8h hi8 = *(const v8h*)&H2[base + 16];
                v16h afrag = __builtin_shufflevector(lo, hi8,
                        0, 1, 2, 3, 4, 5, 6, 7, 8, 9, 10, 11, 12, 13, 14, 15);
                v16h bfrag = *(const v16h*)&W2B[(kc * 32 + lane) * 16];
                acc = __builtin_amdgcn_wmma_f32_16x16x32_f16(
                        false, afrag, false, bfrag, (short)0, acc, false, false);
            }
            // D row = cell, col = digit -> 8 contiguous floats of logitsT[digit]
            float4 s0, s1;
            s0.x = acc[0]; s0.y = acc[1]; s0.z = acc[2]; s0.w = acc[3];
            s1.x = acc[4]; s1.y = acc[5]; s1.z = acc[6]; s1.w = acc[7];
            *(float4*)&logitsT[n16 * LSTR + mt * 16 + hoff]     = s0;
            *(float4*)&logitsT[n16 * LSTR + mt * 16 + hoff + 4] = s1;
        }
        __syncthreads();

        // ---- 4) softmax over digits, x_pred update, packed-key argmax ----
        if (tid < CELLS) {
            const int c = tid;
            float xs = 0.f;
            #pragma unroll
            for (int d = 0; d < 9; ++d) xs += xL[c * 9 + d];
            const bool empty = (xs == 0.f);

            float mx = -1e30f;
            #pragma unroll
            for (int d = 0; d < 9; ++d) mx = fmaxf(mx, logitsT[d * LSTR + c]);
            float p[9], sum = 0.f;
            #pragma unroll
            for (int d = 0; d < 9; ++d) {
                p[d] = __expf(logitsT[d * LSTR + c] - mx);
                sum += p[d];
            }
            const float inv = 1.f / sum;

            if (empty) {
                float vbest = 0.f; int dbest = 0;
                #pragma unroll
                for (int d = 0; d < 9; ++d) {
                    float pd = p[d] * inv;
                    xpredL[c * 9 + d] = pd;
                    if (pd > vbest) { vbest = pd; dbest = d; }   // first-index tie break
                }
                // key: prob bits (non-negative -> monotone as uint) | inverted cell | digit
                unsigned long long key =
                    ((unsigned long long)__float_as_uint(vbest) << 32) |
                    ((unsigned long long)(255 - c) << 8) |
                    (unsigned long long)dbest;
                atomicMax(&bestKey, key);
            }
        }
        __syncthreads();

        // ---- 5) fill the single most confident empty cell ----
        if (tid == 0) {
            unsigned long long key = bestKey;
            if (key != 0ull) {   // key==0 <=> no empty cell
                int c = 255 - (int)((key >> 8) & 0xFF);
                int d = (int)(key & 0xFF);
                float old = xL[c * 9 + d];
                xL[c * 9 + d] = fmaxf(old, 1.f);
            }
            bestKey = 0ull;
        }
        __syncthreads();
    }

    // ---- write outputs: (x_pred, x) concatenated ----
    for (int i = tid; i < BOARD; i += 256) {
        out_xpred[b * BOARD + i] = xpredL[i];
        out_x   [b * BOARD + i] = xL[i];
    }
}

extern "C" void kernel_launch(void* const* d_in, const int* in_sizes, int n_in,
                              void* d_out, int out_size, void* d_ws, size_t ws_size,
                              hipStream_t stream)
{
    (void)in_sizes; (void)n_in; (void)out_size; (void)d_ws; (void)ws_size;
    const float* x   = (const float*)d_in[0];
    // d_in[1] = constraint_mask: row/col/box structure is hardcoded
    const float* W1  = (const float*)d_in[2];
    const float* b1  = (const float*)d_in[3];
    const float* W2  = (const float*)d_in[4];
    const float* b2  = (const float*)d_in[5];
    const int*   nit = (const int*)d_in[6];

    float* out_xpred = (float*)d_out;
    float* out_x     = (float*)d_out + (size_t)BATCH * BOARD;

    sudoku_solver_kernel<<<BATCH, 256, 0, stream>>>(x, W1, b1, W2, b2, nit,
                                                    out_xpred, out_x);
}
